// Mol2SpecEGNN_1864015806924
// MI455X (gfx1250) — matrix operand order repaired and checked by reading, not compiled, not run on hardware
//
#include <hip/hip_runtime.h>
#include <hip/hip_bf16.h>
#include <stdint.h>

// ---------------------------------------------------------------------------
// Types for CDNA5 WMMA (gfx1250, wave32)
// ---------------------------------------------------------------------------
typedef __bf16 bf16_t;
typedef __attribute__((ext_vector_type(16))) __bf16 v16bf;
typedef __attribute__((ext_vector_type(8)))  __bf16 v8bf;
typedef __attribute__((ext_vector_type(8)))  float  v8f;

__device__ __forceinline__ float silu_f(float v) {
    return v * (1.0f / (1.0f + __expf(-v)));
}

// float atomic max via int ordering trick (dest must be initialized to -inf)
__device__ __forceinline__ void atomicMaxFloat(float* addr, float val) {
    if (val >= 0.0f) atomicMax((int*)addr, __float_as_int(val));
    else             atomicMin((unsigned int*)addr, __float_as_uint(val));
}

// CDNA5 async global->LDS copy (16B per lane), tracked by ASYNCcnt.
__device__ __forceinline__ void async_load_b128(uint32_t lds_off, const void* gaddr) {
    asm volatile("global_load_async_to_lds_b128 %0, %1, off"
                 :: "v"(lds_off), "v"(gaddr) : "memory");
}
__device__ __forceinline__ void wait_async() {
    asm volatile("s_wait_asynccnt 0x0" ::: "memory");
}
__device__ __forceinline__ uint32_t lds_addr(const void* p) {
    return (uint32_t)(size_t)p;   // low 32 bits of flat addr = LDS offset
}

// ---------------------------------------------------------------------------
// WMMA fragment loaders (ISA 7.12.2 layouts)
// A 16x32 bf16: lane half h (=lane>>4), row m=lane&15:
//   elems 0..7  -> K = 8*h + 0..7        (contig in LDS row)
//   elems 8..15 -> K = 16 + 8*h + 0..7   (contig in LDS row)
// B 32x16 bf16 staged TRANSPOSED in LDS (row = N column, 32 K elems):
//   elems 0..7 -> K = 8*h + 0..7, elems 8..15 -> K = 16 + 8*h + 0..7
// ---------------------------------------------------------------------------
__device__ __forceinline__ v16bf frag_a(const bf16_t* row, int half) {
    v8bf lo = *(const v8bf*)(row + half * 8);
    v8bf hi = *(const v8bf*)(row + half * 8 + 16);
    return __builtin_shufflevector(lo, hi, 0,1,2,3,4,5,6,7,8,9,10,11,12,13,14,15);
}
__device__ __forceinline__ v16bf frag_b(const bf16_t* row, int half) {
    v8bf lo = *(const v8bf*)(row + half * 8);
    v8bf hi = *(const v8bf*)(row + half * 8 + 16);
    return __builtin_shufflevector(lo, hi, 0,1,2,3,4,5,6,7,8,9,10,11,12,13,14,15);
}
#define WMMA_BF16(a, b, c) \
    __builtin_amdgcn_wmma_f32_16x16x32_bf16(false, a, false, b, (short)0, c, false, false)

#define BM 32
#define BN 256
#define BK 32
#define APITCH 40   // bf16 elems; 80B pitch -> 16B aligned chunks, conflict-skewed

// ---------------------------------------------------------------------------
// Generic bf16 WMMA GEMM:  out = act(A[M,K] @ Bw[K,N] + bias)
// 256 threads = 8 waves. Tile 32x256: wave w owns col strips w*16 and 128+w*16,
// 2 M-tiles each -> 4 WMMA per k-step with full A/B fragment reuse.
// ACT: 0=none 1=silu
// OMODE: 0: store bf16(outBF)
//        2: outF += v; outBF = bf16(outF)   (residual add)
//        3: atomicMax into outF[pool_idx[m]*N + n]  (segment max pool)
//        4: store f32(outF)
// Requires: M%32==0, K%32==0, N%8==0 (tail N-tiles handled by guards).
// ---------------------------------------------------------------------------
template<int ACT, int OMODE>
__global__ __launch_bounds__(256) void gemm_bf16_kernel(
    const bf16_t* __restrict__ A, const bf16_t* __restrict__ Bw,
    const float* __restrict__ bias,
    float* __restrict__ outF, bf16_t* __restrict__ outBF,
    const int* __restrict__ pool_idx,
    int M, int Nn, int K)
{
    __shared__ alignas(16) bf16_t sA[BM][APITCH];
    __shared__ alignas(16) bf16_t sB[BN][APITCH];

    const int tid  = threadIdx.x;
    const int wave = tid >> 5;
    const int lane = tid & 31;
    const int half = lane >> 4;
    const int l15  = lane & 15;
    const int mbase = blockIdx.x * BM;
    const int nbase = blockIdx.y * BN;

    v8f acc[2][2] = {};

    for (int kb = 0; kb < K; kb += BK) {
        // ---- A tile (32x32 bf16): async global->LDS, threads 0..127 ----
        if (tid < 128) {
            int r = tid >> 2;
            int c = (tid & 3) * 8;
            async_load_b128(lds_addr(&sA[r][c]),
                            A + (size_t)(mbase + r) * K + kb + c);
        }
        // ---- B tile (32x256 bf16) transposed into sB[n][k] ----
        #pragma unroll
        for (int it = 0; it < 4; ++it) {
            int cid = tid + it * 256;        // 0..1023
            int k   = cid >> 5;              // 0..31
            int n0  = (cid & 31) * 8;        // 0..248
            uint4 vv = make_uint4(0u, 0u, 0u, 0u);
            if (nbase + n0 < Nn)
                vv = *(const uint4*)(Bw + (size_t)(kb + k) * Nn + nbase + n0);
            const bf16_t* t = (const bf16_t*)&vv;
            #pragma unroll
            for (int i = 0; i < 8; ++i) sB[n0 + i][k] = t[i];
        }
        if (tid == 0 && kb + BK < K)
            __builtin_prefetch(Bw + (size_t)(kb + BK) * Nn + nbase, 0, 1);
        wait_async();
        __syncthreads();

        v16bf a0 = frag_a(&sA[l15][0],      half);
        v16bf a1 = frag_a(&sA[16 + l15][0], half);
        v16bf b0 = frag_b(&sB[wave * 16 + l15][0],       half);
        v16bf b1 = frag_b(&sB[128 + wave * 16 + l15][0], half);
        acc[0][0] = WMMA_BF16(a0, b0, acc[0][0]);
        acc[1][0] = WMMA_BF16(a1, b0, acc[1][0]);
        acc[0][1] = WMMA_BF16(a0, b1, acc[0][1]);
        acc[1][1] = WMMA_BF16(a1, b1, acc[1][1]);
        __syncthreads();
    }

    // ---- epilogue: D layout -> VGPR r holds M = r + 8*half, N = lane&15 ----
    #pragma unroll
    for (int s = 0; s < 2; ++s) {
        int ng = nbase + s * 128 + wave * 16 + l15;
        if (ng >= Nn) continue;
        float bv = bias[ng];
        #pragma unroll
        for (int mb = 0; mb < 2; ++mb) {
            #pragma unroll
            for (int r = 0; r < 8; ++r) {
                int mg = mbase + mb * 16 + r + 8 * half;
                float v = acc[mb][s][r] + bv;
                if (ACT == 1) v = silu_f(v);
                size_t o = (size_t)mg * Nn + ng;
                if (OMODE == 0) {
                    outBF[o] = (bf16_t)v;
                } else if (OMODE == 2) {
                    float sum = outF[o] + v;
                    outF[o]  = sum;
                    outBF[o] = (bf16_t)sum;
                } else if (OMODE == 3) {
                    atomicMaxFloat(&outF[(size_t)pool_idx[mg] * Nn + ng], v);
                } else {
                    outF[o] = v;
                }
            }
        }
    }
}

// ---------------------------------------------------------------------------
// Edge MLP layer-1 GEMM with fused gather/concat A-tile:
//   ein[e] = [ h[row[e]] (512) | h[col[e]] (512) | radial[e] (1) | edge_attr[e] (4) ]
//   out[e] = silu(ein @ W + bias)          K=1029 (padded 1056), N=512
// The k-tile source is uniform per kb (scalar branch): 16 tiles from h[row],
// 16 from h[col] (async 16B gathers), one mixed tail tile handled per-element.
// ---------------------------------------------------------------------------
__global__ __launch_bounds__(256) void edge_mlp1_kernel(
    const bf16_t* __restrict__ hbf,
    const bf16_t* __restrict__ W,       // [1029][512] bf16 (layer slice)
    const float*  __restrict__ bias,    // [512]
    const float*  __restrict__ eattr,   // [E][4]
    const int*    __restrict__ ei,      // [2][E]
    const float*  __restrict__ radial,  // [E]
    bf16_t* __restrict__ out,           // [E][512]
    int Etot)
{
    __shared__ alignas(16) bf16_t sA[BM][APITCH];
    __shared__ alignas(16) bf16_t sB[BN][APITCH];
    __shared__ int   sRow[32], sCol[32];
    __shared__ float sRad[32];

    const int tid  = threadIdx.x;
    const int wave = tid >> 5;
    const int lane = tid & 31;
    const int half = lane >> 4;
    const int l15  = lane & 15;
    const int ebase = blockIdx.x * BM;
    const int nbase = blockIdx.y * BN;
    const int K = 1029, KP = 1056, Nn = 512;

    if (tid < 32) {
        int e = ebase + tid;
        sRow[tid] = ei[e];
        sCol[tid] = ei[Etot + e];
        sRad[tid] = radial[e];
    }
    __syncthreads();

    v8f acc[2][2] = {};

    for (int kb = 0; kb < KP; kb += BK) {
        // ---- A tile: gathered async copies; source uniform per kb ----
        if (kb < 512) {                       // wholly inside h[row] block
            if (tid < 128) {
                int r = tid >> 2, c = (tid & 3) * 8;
                async_load_b128(lds_addr(&sA[r][c]),
                                hbf + (size_t)sRow[r] * 512 + kb + c);
            }
        } else if (kb < 1024) {               // wholly inside h[col] block
            if (tid < 128) {
                int r = tid >> 2, c = (tid & 3) * 8;
                async_load_b128(lds_addr(&sA[r][c]),
                                hbf + (size_t)sCol[r] * 512 + (kb - 512) + c);
            }
        } else {                              // single mixed tail tile (kb==1024)
            #pragma unroll
            for (int j = 0; j < 4; ++j) {
                int lin = tid + j * 256;
                int r = lin >> 5, c = lin & 31;
                int k = kb + c;
                bf16_t v;
                if (k == 1024)     v = (bf16_t)sRad[r];
                else if (k < 1029) v = (bf16_t)eattr[(size_t)(ebase + r) * 4 + (k - 1025)];
                else               v = (bf16_t)0.0f;
                sA[r][c] = v;
            }
        }
        // ---- B tile (zero-padded past K=1029), transposed ----
        #pragma unroll
        for (int it = 0; it < 4; ++it) {
            int cid = tid + it * 256;
            int k   = cid >> 5;
            int n0  = (cid & 31) * 8;
            uint4 vv = make_uint4(0u, 0u, 0u, 0u);
            if (kb + k < K)
                vv = *(const uint4*)(W + (size_t)(kb + k) * Nn + nbase + n0);
            const bf16_t* t = (const bf16_t*)&vv;
            #pragma unroll
            for (int i = 0; i < 8; ++i) sB[n0 + i][k] = t[i];
        }
        if (tid == 0 && kb + BK < K)
            __builtin_prefetch(W + (size_t)(kb + BK) * Nn + nbase, 0, 1);
        wait_async();
        __syncthreads();

        v16bf a0 = frag_a(&sA[l15][0],      half);
        v16bf a1 = frag_a(&sA[16 + l15][0], half);
        v16bf b0 = frag_b(&sB[wave * 16 + l15][0],       half);
        v16bf b1 = frag_b(&sB[128 + wave * 16 + l15][0], half);
        acc[0][0] = WMMA_BF16(a0, b0, acc[0][0]);
        acc[1][0] = WMMA_BF16(a1, b0, acc[1][0]);
        acc[0][1] = WMMA_BF16(a0, b1, acc[0][1]);
        acc[1][1] = WMMA_BF16(a1, b1, acc[1][1]);
        __syncthreads();
    }

    #pragma unroll
    for (int s = 0; s < 2; ++s) {
        int ng = nbase + s * 128 + wave * 16 + l15;
        float bv = bias[ng];
        #pragma unroll
        for (int mb = 0; mb < 2; ++mb) {
            #pragma unroll
            for (int r = 0; r < 8; ++r) {
                int mg = ebase + mb * 16 + r + 8 * half;
                float v = silu_f(acc[mb][s][r] + bv);
                out[(size_t)mg * 512 + ng] = (bf16_t)v;
            }
        }
    }
}

// ---------------------------------------------------------------------------
// Small helper kernels
// ---------------------------------------------------------------------------
__global__ void fill_u32_kernel(uint32_t* p, uint32_t v, size_t n) {
    size_t i = (size_t)blockIdx.x * 256 + threadIdx.x;
    if (i < n) p[i] = v;
}
__global__ void cvt_f32_bf16_kernel(const float* s, bf16_t* d, size_t n) {
    size_t i = (size_t)blockIdx.x * 256 + threadIdx.x;
    if (i < n) d[i] = (bf16_t)s[i];
}
__global__ void copy_f32_kernel(const float* s, float* d, size_t n) {
    size_t i = (size_t)blockIdx.x * 256 + threadIdx.x;
    if (i < n) d[i] = s[i];
}
// h = x @ emb_in_w + b   (K=27, tiny: VALU kernel, block per node)
__global__ __launch_bounds__(256) void embed_kernel(
    const float* __restrict__ x, const float* __restrict__ w,
    const float* __restrict__ b, float* __restrict__ h, bf16_t* __restrict__ hbf)
{
    __shared__ float sx[27];
    int n = blockIdx.x;
    if (threadIdx.x < 27) sx[threadIdx.x] = x[(size_t)n * 27 + threadIdx.x];
    __syncthreads();
    for (int j = threadIdx.x; j < 512; j += 256) {
        float a = b[j];
        #pragma unroll
        for (int k = 0; k < 27; ++k) a = fmaf(sx[k], w[k * 512 + j], a);
        h[(size_t)n * 512 + j]   = a;
        hbf[(size_t)n * 512 + j] = (bf16_t)a;
    }
}
__global__ void deg_kernel(const int* __restrict__ ei, float* __restrict__ deg, int E) {
    int e = blockIdx.x * 256 + threadIdx.x;
    if (e < E) atomicAdd(&deg[ei[e]], 1.0f);
}
__global__ void radial_kernel(const float* __restrict__ pos, const int* __restrict__ ei,
                              float* __restrict__ rad, int E) {
    int e = blockIdx.x * 256 + threadIdx.x;
    if (e >= E) return;
    int r = ei[e], c = ei[E + e];
    float dx = pos[r*3+0] - pos[c*3+0];
    float dy = pos[r*3+1] - pos[c*3+1];
    float dz = pos[r*3+2] - pos[c*3+2];
    rad[e] = dx*dx + dy*dy + dz*dz;
}
// cw[e] = cmid[e,:] . w2 ; acc[row[e]] += diff * cw   (wave per edge)
__global__ __launch_bounds__(256) void coord_scatter_kernel(
    const bf16_t* __restrict__ cmid, const float* __restrict__ w2,
    const int* __restrict__ ei, const float* __restrict__ pos,
    float* __restrict__ acc, int E)
{
    int wave = threadIdx.x >> 5, lane = threadIdx.x & 31;
    int e = blockIdx.x * 8 + wave;
    if (e >= E) return;
    const bf16_t* row = cmid + (size_t)e * 512;
    float s = 0.0f;
    for (int i = lane; i < 512; i += 32) s += (float)row[i] * w2[i];
    #pragma unroll
    for (int off = 16; off > 0; off >>= 1) s += __shfl_down(s, off, 32);
    s = __shfl(s, 0, 32);
    if (lane < 3) {
        int r = ei[e], c = ei[E + e];
        float d = pos[r * 3 + lane] - pos[c * 3 + lane];
        atomicAdd(&acc[r * 3 + lane], d * s);
    }
}
// agg[row[e], k] += m[e, k]
__global__ void agg_kernel(const bf16_t* __restrict__ m, const int* __restrict__ ei,
                           float* __restrict__ agg, int E) {
    size_t idx = (size_t)blockIdx.x * 256 + threadIdx.x;
    if (idx >= (size_t)E * 512) return;
    int e = (int)(idx >> 9), k = (int)(idx & 511);
    atomicAdd(&agg[(size_t)ei[e] * 512 + k], (float)m[idx]);
}
__global__ void pos_update_kernel(float* __restrict__ pos, const float* __restrict__ acc,
                                  const float* __restrict__ deg, int n3) {
    int i = blockIdx.x * 256 + threadIdx.x;
    if (i < n3) pos[i] += acc[i] / fmaxf(deg[i / 3], 1.0f);
}
// nin = [h | agg] as bf16
__global__ void nin_kernel(const bf16_t* __restrict__ hbf, const float* __restrict__ agg,
                           bf16_t* __restrict__ nin, int Nnode) {
    size_t idx = (size_t)blockIdx.x * 256 + threadIdx.x;
    if (idx >= (size_t)Nnode * 1024) return;
    int n = (int)(idx >> 10), k = (int)(idx & 1023);
    nin[idx] = (k < 512) ? hbf[(size_t)n * 512 + k] : (bf16_t)agg[(size_t)n * 512 + (k - 512)];
}
// feat = [g | x1 | frag | adduct]
__global__ void feat_kernel(const float* __restrict__ g, const float* __restrict__ x1,
                            const float* __restrict__ fr, const float* __restrict__ ad,
                            float* __restrict__ feat, bf16_t* __restrict__ featbf, int total) {
    int i = blockIdx.x * 256 + threadIdx.x;
    if (i >= total) return;
    int b = i / 1344, j = i % 1344;
    float v;
    if (j < 1024)      v = g[(size_t)b * 1024 + j];
    else if (j < 1280) v = x1[(size_t)b * 256 + (j - 1024)];
    else if (j < 1312) v = fr[(size_t)b * 32 + (j - 1280)];
    else               v = ad[(size_t)b * 32 + (j - 1312)];
    feat[i]   = v;
    featbf[i] = (bf16_t)v;
}

// ---------------------------------------------------------------------------
// Host orchestration
// ---------------------------------------------------------------------------
extern "C" void kernel_launch(void* const* d_in, const int* in_sizes, int n_in,
                              void* d_out, int out_size, void* d_ws, size_t ws_size,
                              hipStream_t stream) {
    constexpr int kN = 40000, kE = 200000, kB = 256, kL = 4;
    constexpr int HD = 512, EIN = 1029, OUTNF = 1024, HEADIN = 1344, PROP = 1000;

    const float* x        = (const float*)d_in[0];
    const float* pos      = (const float*)d_in[1];
    const float* eattr    = (const float*)d_in[2];
    const float* x1       = (const float*)d_in[3];
    const float* frag     = (const float*)d_in[4];
    const float* adduct   = (const float*)d_in[5];
    const float* emb_in_w = (const float*)d_in[6];
    const float* emb_in_b = (const float*)d_in[7];
    const float* edge_w1  = (const float*)d_in[8];
    const float* edge_b1  = (const float*)d_in[9];
    const float* edge_w2  = (const float*)d_in[10];
    const float* edge_b2  = (const float*)d_in[11];
    const float* node_w1  = (const float*)d_in[12];
    const float* node_b1  = (const float*)d_in[13];
    const float* node_w2  = (const float*)d_in[14];
    const float* node_b2  = (const float*)d_in[15];
    const float* coord_w1 = (const float*)d_in[16];
    const float* coord_b1 = (const float*)d_in[17];
    const float* coord_w2 = (const float*)d_in[18];
    const float* emb_out_w= (const float*)d_in[19];
    const float* emb_out_b= (const float*)d_in[20];
    const float* res_w1   = (const float*)d_in[21];
    const float* res_b1   = (const float*)d_in[22];
    const float* res_w2   = (const float*)d_in[23];
    const float* res_b2   = (const float*)d_in[24];
    const float* head_w   = (const float*)d_in[25];
    const float* head_b   = (const float*)d_in[26];
    const int*   eidx     = (const int*)d_in[27];
    const int*   batch    = (const int*)d_in[28];

    // ---- bump allocator in workspace ----
    char* wp = (char*)d_ws;
    auto alloc = [&](size_t bytes) -> char* {
        char* r = wp; wp += (bytes + 255) & ~(size_t)255; return r;
    };
    float*  h_f   = (float*)alloc((size_t)kN * HD * 4);
    bf16_t* h_bf  = (bf16_t*)alloc((size_t)kN * HD * 2);
    bf16_t* hid   = (bf16_t*)alloc((size_t)kE * HD * 2);   // edge hidden / cmid / node hidden
    bf16_t* m_bf  = (bf16_t*)alloc((size_t)kE * HD * 2);
    float*  agg   = (float*)alloc((size_t)kN * HD * 4);
    bf16_t* nin   = (bf16_t*)alloc((size_t)kN * 1024 * 2);
    float*  rad   = (float*)alloc((size_t)kE * 4);
    float*  deg   = (float*)alloc((size_t)kN * 4);
    float*  posc  = (float*)alloc((size_t)kN * 3 * 4);
    float*  pacc  = (float*)alloc((size_t)kN * 3 * 4);
    float*  g     = (float*)alloc((size_t)kB * OUTNF * 4);
    float*  feat  = (float*)alloc((size_t)kB * HEADIN * 4);
    bf16_t* featbf= (bf16_t*)alloc((size_t)kB * HEADIN * 2);
    bf16_t* r1bf  = (bf16_t*)alloc((size_t)kB * HD * 2);
    bf16_t* f2bf  = (bf16_t*)alloc((size_t)kB * HEADIN * 2);
    bf16_t* w_e1  = (bf16_t*)alloc((size_t)kL * EIN * HD * 2);
    bf16_t* w_e2  = (bf16_t*)alloc((size_t)kL * HD * HD * 2);
    bf16_t* w_n1  = (bf16_t*)alloc((size_t)kL * 1024 * HD * 2);
    bf16_t* w_n2  = (bf16_t*)alloc((size_t)kL * HD * HD * 2);
    bf16_t* w_c1  = (bf16_t*)alloc((size_t)kL * HD * HD * 2);
    bf16_t* w_eo  = (bf16_t*)alloc((size_t)HD * OUTNF * 2);
    bf16_t* w_r1  = (bf16_t*)alloc((size_t)HEADIN * HD * 2);
    bf16_t* w_r2  = (bf16_t*)alloc((size_t)HD * HEADIN * 2);
    bf16_t* w_hd  = (bf16_t*)alloc((size_t)HEADIN * PROP * 2);

    auto blk1 = [](size_t n) { return (unsigned)((n + 255) / 256); };
    auto cvt = [&](const float* s, bf16_t* d, size_t n) {
        cvt_f32_bf16_kernel<<<blk1(n), 256, 0, stream>>>(s, d, n);
    };
    auto fill = [&](void* p, uint32_t v, size_t n) {
        fill_u32_kernel<<<blk1(n), 256, 0, stream>>>((uint32_t*)p, v, n);
    };
#define GEMM(ACT, OMODE, A_, B_, bias_, outF_, outBF_, pool_, M_, N_, K_)          \
    gemm_bf16_kernel<ACT, OMODE><<<dim3((unsigned)((M_) / 32),                     \
        (unsigned)(((N_) + 255) / 256)), 256, 0, stream>>>(                        \
        A_, B_, bias_, outF_, outBF_, pool_, M_, N_, K_)

    // ---- weight conversion to bf16 ----
    cvt(edge_w1,  w_e1, (size_t)kL * EIN * HD);
    cvt(edge_w2,  w_e2, (size_t)kL * HD * HD);
    cvt(node_w1,  w_n1, (size_t)kL * 1024 * HD);
    cvt(node_w2,  w_n2, (size_t)kL * HD * HD);
    cvt(coord_w1, w_c1, (size_t)kL * HD * HD);
    cvt(emb_out_w,w_eo, (size_t)HD * OUTNF);
    cvt(res_w1,   w_r1, (size_t)HEADIN * HD);
    cvt(res_w2,   w_r2, (size_t)HD * HEADIN);
    cvt(head_w,   w_hd, (size_t)HEADIN * PROP);

    // ---- setup: pos copy, degree, embedding ----
    copy_f32_kernel<<<blk1((size_t)kN * 3), 256, 0, stream>>>(pos, posc, (size_t)kN * 3);
    fill(deg, 0, kN);
    deg_kernel<<<blk1(kE), 256, 0, stream>>>(eidx, deg, kE);
    embed_kernel<<<kN, 256, 0, stream>>>(x, emb_in_w, emb_in_b, h_f, h_bf);

    // ---- EGNN layers ----
    for (int l = 0; l < kL; ++l) {
        radial_kernel<<<blk1(kE), 256, 0, stream>>>(posc, eidx, rad, kE);
        // hid = silu(ein @ W1 + b1)   [fused gather/concat, K=1029]
        edge_mlp1_kernel<<<dim3(kE / 32, HD / 256), 256, 0, stream>>>(
            h_bf, w_e1 + (size_t)l * EIN * HD, edge_b1 + l * HD,
            eattr, eidx, rad, hid, kE);
        // m = silu(hid @ W2 + b2)
        GEMM(1, 0, hid, w_e2 + (size_t)l * HD * HD, edge_b2 + l * HD,
             (float*)nullptr, m_bf, (const int*)nullptr, kE, HD, HD);
        // cmid = silu(m @ coord_w1 + cb1)  (reuse hid)
        GEMM(1, 0, m_bf, w_c1 + (size_t)l * HD * HD, coord_b1 + l * HD,
             (float*)nullptr, hid, (const int*)nullptr, kE, HD, HD);
        // coord update: acc[row] += diff * (cmid . coord_w2)
        fill(pacc, 0, (size_t)kN * 3);
        coord_scatter_kernel<<<kE / 8, 256, 0, stream>>>(
            hid, coord_w2 + (size_t)l * HD, eidx, posc, pacc, kE);
        // agg = segment_sum(m, row)
        fill(agg, 0, (size_t)kN * HD);
        agg_kernel<<<blk1((size_t)kE * HD), 256, 0, stream>>>(m_bf, eidx, agg, kE);
        // pos += acc / deg
        pos_update_kernel<<<blk1((size_t)kN * 3), 256, 0, stream>>>(posc, pacc, deg, kN * 3);
        // node update: h += silu([h|agg] @ nw1 + nb1) @ nw2 + nb2
        nin_kernel<<<blk1((size_t)kN * 1024), 256, 0, stream>>>(h_bf, agg, nin, kN);
        GEMM(1, 0, nin, w_n1 + (size_t)l * 1024 * HD, node_b1 + l * HD,
             (float*)nullptr, hid, (const int*)nullptr, kN, HD, 1024);
        GEMM(0, 2, hid, w_n2 + (size_t)l * HD * HD, node_b2 + l * HD,
             h_f, h_bf, (const int*)nullptr, kN, HD, HD);
    }

    // ---- h @ emb_out + b, fused global max pool over batch ----
    fill(g, 0xFF800000u, (size_t)kB * OUTNF);   // -inf
    GEMM(0, 3, h_bf, w_eo, emb_out_b, g, (bf16_t*)nullptr, batch, kN, OUTNF, HD);

    // ---- head: feat -> ResBlock -> linear ----
    feat_kernel<<<blk1((size_t)kB * HEADIN), 256, 0, stream>>>(
        g, x1, frag, adduct, feat, featbf, kB * HEADIN);
    GEMM(1, 0, featbf, w_r1, res_b1, (float*)nullptr, r1bf, (const int*)nullptr,
         kB, HD, HEADIN);
    GEMM(0, 2, r1bf, w_r2, res_b2, feat, f2bf, (const int*)nullptr,
         kB, HEADIN, HD);
    GEMM(0, 4, f2bf, w_hd, head_b, (float*)d_out, (bf16_t*)nullptr, (const int*)nullptr,
         kB, PROP, HEADIN);
#undef GEMM
}
